// MultiHeadAttention_39883066311253
// MI455X (gfx1250) — compile-verified, matrix operand order checked by reading, not run on hardware
//
#include <hip/hip_runtime.h>

typedef _Float16 h16;
typedef __attribute__((ext_vector_type(16))) _Float16 v16h;
typedef __attribute__((ext_vector_type(8)))  float    v8f;
typedef __attribute__((ext_vector_type(4)))  unsigned int u32x4;
typedef __attribute__((ext_vector_type(8)))  unsigned int u32x8;
typedef __attribute__((ext_vector_type(4)))  float    f32x4;

#define NB    2
#define NS    2048
#define NDM   1024
#define NHEAD 8
#define NDH   128

// ---- load a 16x32 f16 A/B fragment from row-major LDS [rows][ldw], K contiguous ----
// ISA 16-bit fragment layout: lanes 0-15 hold row (lane), K {k0..k0+7, k0+16..k0+23};
// lanes 16-31 hold row (lane-16), K {k0+8..k0+15, k0+24..k0+31}. Two b128 DS reads.
__device__ __forceinline__ v16h frag_ld(const h16* lds, int ldw, int row0, int k0) {
  const int lane = threadIdx.x & 31;
  const h16* p = lds + (row0 + (lane & 15)) * ldw + k0 + ((lane >> 4) << 3);
  union { v16h v; u32x4 q[2]; } u;
  u.q[0] = *(const u32x4*)p;
  u.q[1] = *(const u32x4*)(p + 16);
  return u.v;
}

__device__ __forceinline__ v8f wmma16x16x32(v16h a, v16h b, v8f c) {
  return __builtin_amdgcn_wmma_f32_16x16x32_f16(
      /*neg_a=*/false, a, /*neg_b=*/false, b,
      /*c_mod=*/(short)0, c, /*reuse_a=*/false, /*reuse_b=*/false);
}

__device__ __forceinline__ v8f vzero8() {
  v8f z;
  #pragma unroll
  for (int i = 0; i < 8; ++i) z[i] = 0.0f;
  return z;
}

// =====================================================================
// Tensor Data Mover: 2-D tile load, global -> LDS, 2-byte elements, with
// LDS row padding (pad_interval = 64 DWORDs = one 128-elem f16 row,
// pad_amount = 4 DWORDs = 8 f16) so rows land at a 136-element pitch.
// D# packing per CDNA5 ISA 8.3/8.4; issued wave-uniform from SGPRs.
// =====================================================================
__device__ __forceinline__ void tdm_load_2d(unsigned lds_off, const void* gaddr,
                                            unsigned tensor_d0, unsigned tensor_d1,
                                            unsigned tile_d0, unsigned tile_d1,
                                            unsigned stride0)
{
  const unsigned long long ga = (unsigned long long)(size_t)gaddr;
  u32x4 g0;
  g0[0] = 1u;                                            // count=1, user desc
  g0[1] = lds_off;                                       // lds_addr (bytes)
  g0[2] = (unsigned)(ga & 0xffffffffu);                  // global_addr[31:0]
  g0[3] = (unsigned)((ga >> 32) & 0x01ffffffu)           // global_addr[56:32]
        | (2u << 30);                                    // type = 2 (image)
  u32x8 g1;
  g1[0] = (1u << 16)                                     // data_size = 2 bytes
        | (1u << 20)                                     // pad_enable
        | (5u << 22)                                     // pad_interval: 64 DW
        | (3u << 25);                                    // pad_amount: 4 DW
  g1[1] = (tensor_d0 & 0xffffu) << 16;                   // tensor_dim0 lo
  g1[2] = ((tensor_d0 >> 16) & 0xffffu)                  // tensor_dim0 hi
        | ((tensor_d1 & 0xffffu) << 16);                 // tensor_dim1 lo
  g1[3] = ((tensor_d1 >> 16) & 0xffffu)                  // tensor_dim1 hi
        | ((tile_d0 & 0xffffu) << 16);                   // tile_dim0
  g1[4] = (tile_d1 & 0xffffu);                           // tile_dim1, tile_dim2=0
  g1[5] = stride0;                                       // tensor_dim0_stride lo32
  g1[6] = 0u;
  g1[7] = 0u;
  u32x4 gz;
  gz[0] = 0u; gz[1] = 0u; gz[2] = 0u; gz[3] = 0u;        // 2-D: groups 2/3 unused
  asm volatile("tensor_load_to_lds %0, %1, %2, %3"
               :: "s"(g0), "s"(g1), "s"(gz), "s"(gz)
               : "memory");
}

// =====================================================================
// Weight convert + transpose: fp32 W[K][N] -> f16 Wt[N][K]
// =====================================================================
__global__ __launch_bounds__(256)
void cvt_wt_kernel(const float* __restrict__ W, h16* __restrict__ Wt, int K, int N) {
  __shared__ h16 tile[32][33];
  const int n0 = blockIdx.x * 32;
  const int k0 = blockIdx.y * 32;
  const int tx = threadIdx.x & 31;
  const int ty = threadIdx.x >> 5;     // 0..7
  #pragma unroll
  for (int i = 0; i < 32; i += 8)
    tile[ty + i][tx] = (h16)W[(size_t)(k0 + ty + i) * N + n0 + tx];
  __syncthreads();
  #pragma unroll
  for (int i = 0; i < 32; i += 8)
    Wt[(size_t)(n0 + ty + i) * K + k0 + tx] = tile[tx][ty + i];
}

// =====================================================================
// Projection GEMM: fp32 X[4096][1024] @ f16 Wt[n][k] (+bias) -> f16,
// scattered into per-head layout. vmode==0: [b][h][s][d]; vmode==1: [b][h][d][s].
// 128x128 tile, 8 waves, wave = 64x32 patch = 4x2 WMMA accumulators.
// =====================================================================
__global__ __launch_bounds__(256)
void proj_gemm_kernel(const float* __restrict__ X, const h16* __restrict__ Wt,
                      const float* __restrict__ bias, h16* __restrict__ Out,
                      int vmode)
{
  __shared__ __align__(16) h16 As[128][40];
  __shared__ __align__(16) h16 Bs[128][40];
  const int tid  = threadIdx.x;
  const int lane = tid & 31;
  const int wave = tid >> 5;
  const int wm = (wave >> 2) * 64;
  const int wn = (wave & 3) * 32;
  const int m0 = blockIdx.y * 128;
  const int n0 = blockIdx.x * 128;

  v8f acc[4][2];
  #pragma unroll
  for (int mi = 0; mi < 4; ++mi)
    #pragma unroll
    for (int ni = 0; ni < 2; ++ni) acc[mi][ni] = vzero8();

  for (int k0 = 0; k0 < NDM; k0 += 32) {
    __syncthreads();
    { // A tile: 128x32 fp32 -> f16 LDS (coalesced b128 global reads)
      const int r = tid >> 3;
      const int c = (tid & 7) * 4;
      #pragma unroll
      for (int rr = 0; rr < 128; rr += 32) {
        f32x4 f = *(const f32x4*)(X + (size_t)(m0 + r + rr) * NDM + k0 + c);
        As[r + rr][c + 0] = (h16)f[0];
        As[r + rr][c + 1] = (h16)f[1];
        As[r + rr][c + 2] = (h16)f[2];
        As[r + rr][c + 3] = (h16)f[3];
      }
    }
    // B tile: 128 rows x 32 f16, already K-contiguous in Wt
    for (int idx = tid; idx < 512; idx += 256) {
      const int r = idx >> 2, seg = (idx & 3) * 8;
      *(u32x4*)&Bs[r][seg] = *(const u32x4*)(Wt + (size_t)(n0 + r) * NDM + k0 + seg);
    }
    if (k0 + 32 < NDM)  // hint next A tile into cache (global_prefetch_b8)
      __builtin_prefetch(X + (size_t)(m0 + (tid >> 3)) * NDM + k0 + 32, 0, 0);
    __syncthreads();

    v16h a[4];
    #pragma unroll
    for (int mi = 0; mi < 4; ++mi) a[mi] = frag_ld(&As[0][0], 40, wm + mi * 16, 0);
    #pragma unroll
    for (int ni = 0; ni < 2; ++ni) {
      v16h b = frag_ld(&Bs[0][0], 40, wn + ni * 16, 0);
      #pragma unroll
      for (int mi = 0; mi < 4; ++mi)
        acc[mi][ni] = wmma16x16x32(a[mi], b, acc[mi][ni]);
    }
  }

  // Epilogue: C layout -> lane&15 = column, VGPR r = row (+8 for upper lanes)
  const int colb   = lane & 15;
  const int rowsel = (lane >> 4) * 8;
  #pragma unroll
  for (int mi = 0; mi < 4; ++mi)
    #pragma unroll
    for (int ni = 0; ni < 2; ++ni) {
      const int col = n0 + wn + ni * 16 + colb;
      const float bv = bias[col];
      const int rbase = m0 + wm + mi * 16 + rowsel;
      const int hh = col & 7, dd = col >> 3;   // reshape(b,s,D,H): head is fast dim
      #pragma unroll
      for (int r = 0; r < 8; ++r) {
        const int row = rbase + r;
        const int bb = row >> 11, ss = row & 2047;
        const float v = acc[mi][ni][r] + bv;
        const size_t o = vmode
            ? ((size_t)((bb * NHEAD + hh) * NDH + dd) * NS + ss)   // V: [b][h][d][s]
            : ((size_t)((bb * NHEAD + hh) * NS + ss) * NDH + dd);  // Q/K: [b][h][s][d]
        Out[o] = (h16)v;
      }
    }
}

// =====================================================================
// Fused flash attention: one WG per (b, h, 128-query tile).
// Qs/Ks: [row][d] f16; Vts: [d][key] f16; Ps: wave-private P staging.
// Q/K/V tiles are staged by the Tensor Data Mover (tensor_load_to_lds)
// with hardware row padding to the 136-element LDS pitch; waves only do
// WMMA + softmax. Dynamic LDS: 4 * 128*136*2B = 136 KB.
// =====================================================================
#define QS_OFF  0u
#define KS_OFF  34816u      // 128*136*2
#define VTS_OFF 69632u      // 2*128*136*2
#define PS_OFF  104448u     // 3*128*136*2

__global__ __launch_bounds__(256)
void attn_kernel(const h16* __restrict__ qp, const h16* __restrict__ kp,
                 const h16* __restrict__ vp, const int* __restrict__ mask,
                 h16* __restrict__ ao)
{
  extern __shared__ __align__(16) h16 smem[];
  h16* Qs  = smem;                 // [128][136]
  h16* Ks  = Qs  + 128 * 136;     // [128 key][136]
  h16* Vts = Ks  + 128 * 136;     // [128 d][136 key]
  h16* Ps  = Vts + 128 * 136;     // [128 q][136 key]

  const int tid  = threadIdx.x;
  const int lane = tid & 31;
  const int wave = tid >> 5;
  const int blk = blockIdx.x;
  const int qt = blk & 15;
  const int h  = (blk >> 4) & 7;
  const int b  = blk >> 7;

  const h16* qbase = qp + ((size_t)(b * NHEAD + h) * NS + qt * 128) * NDH;
  const h16* kbase = kp + (size_t)(b * NHEAD + h) * NS * NDH;
  const h16* vbase = vp + (size_t)(b * NHEAD + h) * NDH * NS;

  // Q tile via TDM (tensor [2048 rows][128], tile 128x128 at qbase)
  if (wave == 0) {
    tdm_load_2d(QS_OFF, qbase, NDH, NS, NDH, 128, NDH);
    __builtin_amdgcn_s_wait_tensorcnt(0);
  }
  __syncthreads();

  v16h qf[4];
  #pragma unroll
  for (int dc = 0; dc < 4; ++dc) qf[dc] = frag_ld(Qs, 136, wave * 16, dc * 32);

  const int rowsel = (lane >> 4) * 8;
  const int q0 = qt * 128 + wave * 16 + rowsel;     // query row for VGPR r = q0 + r
  const float inv_scale = 0.088388347648318447f;    // 1/sqrt(128)

  float mrow[8], lrow[8];
  v8f out[8];
  #pragma unroll
  for (int r = 0; r < 8; ++r) { mrow[r] = -3.0e38f; lrow[r] = 0.0f; }
  #pragma unroll
  for (int dn = 0; dn < 8; ++dn) out[dn] = vzero8();

  for (int kt = 0; kt < 16; ++kt) {
    __syncthreads();
    if (wave == 0) {
      // K tile: rows = keys (d-contiguous); V tile: rows = d (key-contiguous)
      tdm_load_2d(KS_OFF,  kbase + (size_t)(kt * 128) * NDH, NDH, NS, NDH, 128, NDH);
      tdm_load_2d(VTS_OFF, vbase + (size_t)(kt * 128),       NS, NDH, 128, 128, NS);
      __builtin_amdgcn_s_wait_tensorcnt(0);
    }
    __syncthreads();

    // S = Q K^T  (B-fragment columns = keys; K rows are d-contiguous in LDS)
    v8f sc[8];
    #pragma unroll
    for (int nk = 0; nk < 8; ++nk) {
      v8f s = vzero8();
      #pragma unroll
      for (int dc = 0; dc < 4; ++dc) {
        v16h kf = frag_ld(Ks, 136, nk * 16, dc * 32);
        s = wmma16x16x32(qf[dc], kf, s);
      }
      sc[nk] = s;
    }

    // scale + mask (C layout: key along lane&15, query along VGPR index)
    #pragma unroll
    for (int nk = 0; nk < 8; ++nk) {
      const int key = kt * 128 + nk * 16 + (lane & 15);
      #pragma unroll
      for (int r = 0; r < 8; ++r) {
        const int mv = mask[(size_t)(b * NS + q0 + r) * NS + key];
        const float v = sc[nk][r] * inv_scale;
        sc[nk][r] = (mv == 0) ? -1.0e10f : v;
      }
    }

    // online softmax: row reductions across the 16-lane half-groups
    #pragma unroll
    for (int r = 0; r < 8; ++r) {
      float v = sc[0][r];
      #pragma unroll
      for (int nk = 1; nk < 8; ++nk) v = fmaxf(v, sc[nk][r]);
      #pragma unroll
      for (int off = 1; off < 16; off <<= 1) v = fmaxf(v, __shfl_xor(v, off, 32));
      const float mnew = fmaxf(mrow[r], v);
      const float corr = __expf(mrow[r] - mnew);
      mrow[r] = mnew;
      float rs = 0.0f;
      #pragma unroll
      for (int nk = 0; nk < 8; ++nk) {
        const float p = __expf(sc[nk][r] - mnew);
        sc[nk][r] = p;
        rs += p;
      }
      #pragma unroll
      for (int off = 1; off < 16; off <<= 1) rs += __shfl_xor(rs, off, 32);
      lrow[r] = lrow[r] * corr + rs;
      #pragma unroll
      for (int dn = 0; dn < 8; ++dn) out[dn][r] *= corr;
    }

    // P: C layout -> A-fragment layout via wave-private LDS staging
    #pragma unroll
    for (int nk = 0; nk < 8; ++nk) {
      const int col = nk * 16 + (lane & 15);
      #pragma unroll
      for (int r = 0; r < 8; ++r)
        Ps[(wave * 16 + rowsel + r) * 136 + col] = (h16)sc[nk][r];
    }

    // O += P V  (B-fragment columns = d; Vts rows are key-contiguous)
    v16h pf[4];
    #pragma unroll
    for (int kc = 0; kc < 4; ++kc) pf[kc] = frag_ld(Ps, 136, wave * 16, kc * 32);
    #pragma unroll
    for (int dn = 0; dn < 8; ++dn)
      #pragma unroll
      for (int kc = 0; kc < 4; ++kc) {
        v16h vf = frag_ld(Vts, 136, dn * 16, kc * 32);
        out[dn] = wmma16x16x32(pf[kc], vf, out[dn]);
      }
  }

  // write ao[b][q][d*H + h] (reference re-interleaves head as fast dim)
  #pragma unroll
  for (int dn = 0; dn < 8; ++dn) {
    const int d = dn * 16 + (lane & 15);
    const int c = d * NHEAD + h;
    #pragma unroll
    for (int r = 0; r < 8; ++r) {
      const float v = out[dn][r] / lrow[r];
      ao[(size_t)(b * NS + q0 + r) * NDM + c] = (h16)v;
    }
  }
}

// =====================================================================
// Output GEMM: f16 ao[4096][1024] @ f16 Wo^T (+bo) -> fp32 out
// =====================================================================
__global__ __launch_bounds__(256)
void out_gemm_kernel(const h16* __restrict__ A, const h16* __restrict__ Wt,
                     const float* __restrict__ bias, float* __restrict__ Out)
{
  __shared__ __align__(16) h16 As[128][40];
  __shared__ __align__(16) h16 Bs[128][40];
  const int tid  = threadIdx.x;
  const int lane = tid & 31;
  const int wave = tid >> 5;
  const int wm = (wave >> 2) * 64;
  const int wn = (wave & 3) * 32;
  const int m0 = blockIdx.y * 128;
  const int n0 = blockIdx.x * 128;

  v8f acc[4][2];
  #pragma unroll
  for (int mi = 0; mi < 4; ++mi)
    #pragma unroll
    for (int ni = 0; ni < 2; ++ni) acc[mi][ni] = vzero8();

  for (int k0 = 0; k0 < NDM; k0 += 32) {
    __syncthreads();
    for (int idx = tid; idx < 512; idx += 256) {
      const int r = idx >> 2, seg = (idx & 3) * 8;
      *(u32x4*)&As[r][seg] = *(const u32x4*)(A  + (size_t)(m0 + r) * NDM + k0 + seg);
      *(u32x4*)&Bs[r][seg] = *(const u32x4*)(Wt + (size_t)(n0 + r) * NDM + k0 + seg);
    }
    __syncthreads();

    v16h a[4];
    #pragma unroll
    for (int mi = 0; mi < 4; ++mi) a[mi] = frag_ld(&As[0][0], 40, wm + mi * 16, 0);
    #pragma unroll
    for (int ni = 0; ni < 2; ++ni) {
      v16h b = frag_ld(&Bs[0][0], 40, wn + ni * 16, 0);
      #pragma unroll
      for (int mi = 0; mi < 4; ++mi)
        acc[mi][ni] = wmma16x16x32(a[mi], b, acc[mi][ni]);
    }
  }

  const int colb   = lane & 15;
  const int rowsel = (lane >> 4) * 8;
  #pragma unroll
  for (int mi = 0; mi < 4; ++mi)
    #pragma unroll
    for (int ni = 0; ni < 2; ++ni) {
      const int col = n0 + wn + ni * 16 + colb;
      const float bv = bias[col];
      const int rbase = m0 + wm + mi * 16 + rowsel;
      #pragma unroll
      for (int r = 0; r < 8; ++r)
        Out[(size_t)(rbase + r) * NDM + col] = acc[mi][ni][r] + bv;
    }
}

// =====================================================================
extern "C" void kernel_launch(void* const* d_in, const int* in_sizes, int n_in,
                              void* d_out, int out_size, void* d_ws, size_t ws_size,
                              hipStream_t stream)
{
  const float* q    = (const float*)d_in[0];
  const float* k    = (const float*)d_in[1];
  const float* v    = (const float*)d_in[2];
  const int*   mask = (const int*)d_in[3];
  const float* Wq   = (const float*)d_in[4];
  const float* bq   = (const float*)d_in[5];
  const float* Wk   = (const float*)d_in[6];
  const float* bk   = (const float*)d_in[7];
  const float* Wv   = (const float*)d_in[8];
  const float* bv   = (const float*)d_in[9];
  const float* Wo   = (const float*)d_in[10];
  const float* bo   = (const float*)d_in[11];
  float* out = (float*)d_out;

  h16* ws = (h16*)d_ws;
  const size_t proj_elems = (size_t)NB * NHEAD * NS * NDH;  // 4,194,304
  const size_t w_elems    = (size_t)NDM * NDM;              // 1,048,576
  h16* qp  = ws;
  h16* kp  = qp  + proj_elems;
  h16* vp  = kp  + proj_elems;
  h16* ao  = vp  + proj_elems;
  h16* Wqt = ao  + proj_elems;
  h16* Wkt = Wqt + w_elems;
  h16* Wvt = Wkt + w_elems;
  h16* Wot = Wvt + w_elems;    // total ~40 MB of workspace

  dim3 cgrid(NDM / 32, NDM / 32);
  cvt_wt_kernel<<<cgrid, 256, 0, stream>>>(Wq, Wqt, NDM, NDM);
  cvt_wt_kernel<<<cgrid, 256, 0, stream>>>(Wk, Wkt, NDM, NDM);
  cvt_wt_kernel<<<cgrid, 256, 0, stream>>>(Wv, Wvt, NDM, NDM);
  cvt_wt_kernel<<<cgrid, 256, 0, stream>>>(Wo, Wot, NDM, NDM);

  dim3 ggrid(NDM / 128, (NB * NS) / 128);   // (8, 32)
  proj_gemm_kernel<<<ggrid, 256, 0, stream>>>(q, Wqt, bq, qp, 0);
  proj_gemm_kernel<<<ggrid, 256, 0, stream>>>(k, Wkt, bk, kp, 0);
  proj_gemm_kernel<<<ggrid, 256, 0, stream>>>(v, Wvt, bv, vp, 1);

  const size_t attn_lds = (size_t)4 * 128 * 136 * sizeof(h16);  // 136 KB dynamic LDS
  attn_kernel<<<NB * NHEAD * (NS / 128), 256, attn_lds, stream>>>(qp, kp, vp, mask, ao);

  out_gemm_kernel<<<ggrid, 256, 0, stream>>>(ao, Wot, bo, out);

  (void)in_sizes; (void)n_in; (void)out_size; (void)ws_size;
}
